// ESCNet_8022998909199
// MI455X (gfx1250) — matrix operand order repaired and checked by reading, not compiled, not run on hardware
//
#include <hip/hip_runtime.h>
#include <hip/hip_bf16.h>

typedef __attribute__((ext_vector_type(16))) __bf16 v16bf;
typedef __attribute__((ext_vector_type(8)))  float  v8f;

#define B_   4
#define C_   16
#define H_   256
#define W_   256
#define N_   (H_*W_)
#define S_   100
#define SP_  112            // S padded to multiple of 16
#define KSLICES 128
#define SLICE_N (N_/KSLICES) // 512
#define EPS_ 1e-32f

// ---------------------------------------------------------------------------
// K0: convert pf (f32) -> bf16 for WMMA B operand; zero split-K accumulators.
// ---------------------------------------------------------------------------
__global__ void k_init(const float* __restrict__ pf, __bf16* __restrict__ pf_bf,
                       float* __restrict__ P, float* __restrict__ qsum)
{
    int id = blockIdx.x * blockDim.x + threadIdx.x;
    if (id < B_*C_*N_)    pf_bf[id] = (__bf16)pf[id];
    if (id < B_*2*SP_*C_) P[id]     = 0.f;
    if (id < B_*2*SP_)    qsum[id]  = 0.f;
}

// ---------------------------------------------------------------------------
// K1: P[b,j,t,c] = sum_n Q[b,j,t,n] * pf[b,c,n]  (split-K over n, atomics)
//     qsum[b,j,t] = sum_n Q[b,j,t,n]
// One wave per (slice, t-tile, b, j). A = Q (16t x 32n), B = pf (32n x 16c).
// FULL=true: all 16 rows valid (t-tiles 0..5), no masking in the hot loop.
// FULL=false: tail tile (rows 96..111), rows >= 100 masked to zero.
// ---------------------------------------------------------------------------
template<bool FULL>
__global__ __launch_bounds__(32) void k_pool(
    const float* __restrict__ Q1, const float* __restrict__ Q2,
    const __bf16* __restrict__ pf_bf,
    float* __restrict__ P, float* __restrict__ qsum, int ttBase)
{
    const int lane  = threadIdx.x;
    const int slice = blockIdx.x;
    const int tt    = ttBase + blockIdx.y;
    const int bj    = blockIdx.z;
    const int b     = bj >> 1;

    const float* Q = ((bj & 1) ? Q2 : Q1) + (size_t)b * S_ * N_;
    const int   tRow = tt*16 + (lane & 15);
    const float m    = (FULL || tRow < S_) ? 1.f : 0.f;   // mask padded rows
    const int   tClamp = FULL ? tRow : ((tRow < S_) ? tRow : 0);
    const float*  qrow = Q + (size_t)tClamp * N_ + ((lane >> 4) * 16);
    const __bf16* prow = pf_bf + ((size_t)b*C_ + (lane & 15)) * N_ + ((lane >> 4) * 16);
    const int n0 = slice * SLICE_N;

    v8f acc = {0.f,0.f,0.f,0.f,0.f,0.f,0.f,0.f};
    float qacc = 0.f;

    for (int i = 0; i < SLICE_N/32; ++i) {
        const int n = n0 + i*32;
        float4 f0 = *(const float4*)(qrow + n);
        float4 f1 = *(const float4*)(qrow + n + 4);
        float4 g0 = *(const float4*)(qrow + n + 8);
        float4 g1 = *(const float4*)(qrow + n + 12);
        qacc += ((f0.x+f0.y) + (f0.z+f0.w)) + ((f1.x+f1.y) + (f1.z+f1.w))
              + ((g0.x+g0.y) + (g0.z+g0.w)) + ((g1.x+g1.y) + (g1.z+g1.w));
        v16bf a;
        if (FULL) {
            a[0]=(__bf16)f0.x;  a[1]=(__bf16)f0.y;  a[2]=(__bf16)f0.z;  a[3]=(__bf16)f0.w;
            a[4]=(__bf16)f1.x;  a[5]=(__bf16)f1.y;  a[6]=(__bf16)f1.z;  a[7]=(__bf16)f1.w;
            a[8]=(__bf16)g0.x;  a[9]=(__bf16)g0.y;  a[10]=(__bf16)g0.z; a[11]=(__bf16)g0.w;
            a[12]=(__bf16)g1.x; a[13]=(__bf16)g1.y; a[14]=(__bf16)g1.z; a[15]=(__bf16)g1.w;
        } else {
            a[0]=(__bf16)(f0.x*m);  a[1]=(__bf16)(f0.y*m);  a[2]=(__bf16)(f0.z*m);  a[3]=(__bf16)(f0.w*m);
            a[4]=(__bf16)(f1.x*m);  a[5]=(__bf16)(f1.y*m);  a[6]=(__bf16)(f1.z*m);  a[7]=(__bf16)(f1.w*m);
            a[8]=(__bf16)(g0.x*m);  a[9]=(__bf16)(g0.y*m);  a[10]=(__bf16)(g0.z*m); a[11]=(__bf16)(g0.w*m);
            a[12]=(__bf16)(g1.x*m); a[13]=(__bf16)(g1.y*m); a[14]=(__bf16)(g1.z*m); a[15]=(__bf16)(g1.w*m);
        }
        v16bf bm = *(const v16bf*)(prow + n);
        acc = __builtin_amdgcn_wmma_f32_16x16x32_bf16(
                  false, a, false, bm, (short)0, acc, false, false);
    }

    float* Pp = P + ((size_t)bj * SP_ + tt*16) * C_;
    #pragma unroll
    for (int v = 0; v < 8; ++v) {
        int tr = v + ((lane >= 16) ? 8 : 0);      // D: M = v (+8 for hi lanes)
        atomicAdd(Pp + tr * C_ + (lane & 15), acc[v]);
    }
    atomicAdd(qsum + (size_t)bj * SP_ + tt*16 + (lane & 15), qacc * m);
}

// ---------------------------------------------------------------------------
// K2: rels[b,j,s,t] = exp(-sum_c (spf[c,s]-spf[c,t])^2)   (OMEGA2 == 1.0)
// ---------------------------------------------------------------------------
__global__ void k_rels(const float* __restrict__ spf1, const float* __restrict__ spf2,
                       float* __restrict__ rels)
{
    int id = blockIdx.x * blockDim.x + threadIdx.x;
    if (id >= B_*2*S_*S_) return;
    int t  = id % S_;
    int s  = (id / S_) % S_;
    int bj = id / (S_*S_);
    const float* spf  = (bj & 1) ? spf2 : spf1;
    const float* base = spf + (size_t)(bj >> 1) * C_ * S_;
    float acc = 0.f;
    #pragma unroll
    for (int c = 0; c < C_; ++c) {
        float d = base[c*S_ + s] - base[c*S_ + t];
        acc += d * d;
    }
    rels[id] = __expf(-acc);
}

// ---------------------------------------------------------------------------
// K3: num[c,s] = sum_t rels[s,t]*P[t,c]; den[s] = sum_t rels[s,t]*qsum[t];
//     spfn[b,j,c,s] = num/(den+EPS), zero-padded to SP_.
// ---------------------------------------------------------------------------
__global__ void k_spfn(const float* __restrict__ rels, const float* __restrict__ P,
                       const float* __restrict__ qsum, float* __restrict__ spfn)
{
    int id = blockIdx.x * blockDim.x + threadIdx.x;
    if (id >= B_*2*C_*SP_) return;
    int sp = id % SP_;
    int c  = (id / SP_) % C_;
    int bj = id / (SP_*C_);
    float* dst = spfn + ((size_t)bj*C_ + c)*SP_ + sp;
    if (sp >= S_) { *dst = 0.f; return; }
    const float* r  = rels + (size_t)bj*S_*S_ + (size_t)sp*S_;
    const float* Pp = P    + (size_t)bj*SP_*C_;
    const float* qs = qsum + (size_t)bj*SP_;
    float numv = 0.f, denv = 0.f;
    for (int t = 0; t < S_; ++t) {
        float wv = r[t];
        numv = fmaf(wv, Pp[t*C_ + c], numv);
        denv = fmaf(wv, qs[t], denv);
    }
    *dst = numv / (denv + EPS_);
}

// ---------------------------------------------------------------------------
// K4: pf_sp[b,c,n] = sum_s spfn1[c,s]*Q1[s,n] + sum_s spfn2[c,s]*Q2[s,n]
// Q tile staged transposed in LDS as bf16 so B operand is contiguous per lane.
// 8 waves/block, each wave owns one 16-column group of a 128-wide n tile.
// ---------------------------------------------------------------------------
__global__ __launch_bounds__(256) void k_unpool(
    const float* __restrict__ Q1, const float* __restrict__ Q2,
    const float* __restrict__ spfn, float* __restrict__ pf_sp)
{
    __shared__ __align__(16) __bf16 qtile[2][128][SP_];
    const int b   = blockIdx.y;
    const int n0  = blockIdx.x * 128;
    const int tid = threadIdx.x;

    for (int idx = tid; idx < 2*SP_*128; idx += 256) {
        int j = idx / (SP_*128);
        int r = idx - j*(SP_*128);
        int s = r >> 7;
        int n = r & 127;
        const float* Q = j ? Q2 : Q1;
        float v = (s < S_) ? Q[((size_t)b*S_ + s)*N_ + n0 + n] : 0.f;
        qtile[j][n][s] = (__bf16)v;          // transposed: [n][s]
    }
    __syncthreads();

    const int lane = tid & 31;
    const int w    = tid >> 5;
    v8f acc = {0.f,0.f,0.f,0.f,0.f,0.f,0.f,0.f};

    #pragma unroll
    for (int j = 0; j < 2; ++j) {
        const float* A = spfn + (((size_t)(b*2 + j))*C_ + (lane & 15)) * SP_;
        #pragma unroll
        for (int s0 = 0; s0 < SP_; s0 += 32) {
            int ka = s0 + (lane >> 4) * 16;
            v16bf a;
            #pragma unroll
            for (int h = 0; h < 8; ++h) {
                a[h]     = (__bf16)A[ka + h];
                a[8 + h] = (__bf16)A[ka + 8 + h];
            }
            const __bf16* Bp = &qtile[j][w*16 + (lane & 15)][s0 + ((lane >> 4) * 16)];
            v16bf bm = *(const v16bf*)Bp;
            acc = __builtin_amdgcn_wmma_f32_16x16x32_bf16(
                      false, a, false, bm, (short)0, acc, false, false);
        }
    }

    const int n = n0 + w*16 + (lane & 15);
    #pragma unroll
    for (int v = 0; v < 8; ++v) {
        int c = v + ((lane >= 16) ? 8 : 0);
        pf_sp[((size_t)b*C_ + c)*N_ + n] = acc[v];
    }
}

// ---------------------------------------------------------------------------
// K5: 3x3 conv, 16 -> 2 channels, SAME zero padding.
// ---------------------------------------------------------------------------
__global__ __launch_bounds__(256) void k_conv(
    const float* __restrict__ pf_sp, const float* __restrict__ w_ds,
    const float* __restrict__ b_ds, float* __restrict__ out)
{
    __shared__ float wl[2*C_*9];
    __shared__ float bl[2];
    if (threadIdx.x < 2*C_*9) wl[threadIdx.x] = w_ds[threadIdx.x];
    if (threadIdx.x < 2)      bl[threadIdx.x] = b_ds[threadIdx.x];
    __syncthreads();

    int id = blockIdx.x * 256 + threadIdx.x;      // exactly B_*N_ threads
    int x = id & (W_-1);
    int y = (id >> 8) & (H_-1);
    int b = id >> 16;
    float a0 = bl[0], a1 = bl[1];
    for (int c = 0; c < C_; ++c) {
        const float* pch = pf_sp + ((size_t)b*C_ + c)*N_;
        #pragma unroll
        for (int kh = -1; kh <= 1; ++kh) {
            int yy = y + kh;
            if ((unsigned)yy >= (unsigned)H_) continue;
            #pragma unroll
            for (int kw = -1; kw <= 1; ++kw) {
                int xx = x + kw;
                if ((unsigned)xx >= (unsigned)W_) continue;
                float v  = pch[yy*W_ + xx];
                int   wi = (c*3 + (kh+1))*3 + (kw+1);
                a0 = fmaf(v, wl[wi],          a0);
                a1 = fmaf(v, wl[C_*9 + wi],   a1);
            }
        }
    }
    out[((size_t)b*2    )*N_ + y*W_ + x] = a0;
    out[((size_t)b*2 + 1)*N_ + y*W_ + x] = a1;
}

// ---------------------------------------------------------------------------
extern "C" void kernel_launch(void* const* d_in, const int* in_sizes, int n_in,
                              void* d_out, int out_size, void* d_ws, size_t ws_size,
                              hipStream_t stream)
{
    (void)in_sizes; (void)n_in; (void)out_size; (void)ws_size;
    const float* pf   = (const float*)d_in[0];
    const float* spf1 = (const float*)d_in[1];
    const float* spf2 = (const float*)d_in[2];
    const float* Q1   = (const float*)d_in[3];
    const float* Q2   = (const float*)d_in[4];
    const float* w_ds = (const float*)d_in[5];
    const float* b_ds = (const float*)d_in[6];
    float* out = (float*)d_out;

    char* ws = (char*)d_ws;                       // ~24.5 MB used
    __bf16* pf_bf = (__bf16*)(ws);                //  8,388,608 B
    float*  P     = (float*)(ws + 8388608);       //     57,344 B
    float*  qsum  = (float*)(ws + 8445952);       //      3,584 B
    float*  rels  = (float*)(ws + 8449536);       //    320,000 B
    float*  spfn  = (float*)(ws + 8769536);       //     57,344 B
    float*  pf_sp = (float*)(ws + 8826880);       // 16,777,216 B

    k_init        <<<(B_*C_*N_ + 255)/256, 256, 0, stream>>>(pf, pf_bf, P, qsum);
    // t-tiles 0..5: all rows valid, maskless hot loop
    k_pool<true>  <<<dim3(KSLICES, 6, B_*2), 32, 0, stream>>>(Q1, Q2, pf_bf, P, qsum, 0);
    // t-tile 6: rows 96..111, rows >= 100 masked
    k_pool<false> <<<dim3(KSLICES, 1, B_*2), 32, 0, stream>>>(Q1, Q2, pf_bf, P, qsum, 6);
    k_rels        <<<(B_*2*S_*S_ + 255)/256, 256, 0, stream>>>(spf1, spf2, rels);
    k_spfn        <<<(B_*2*C_*SP_ + 255)/256, 256, 0, stream>>>(rels, P, qsum, spfn);
    k_unpool      <<<dim3(N_/128, B_), 256, 0, stream>>>(Q1, Q2, spfn, pf_sp);
    k_conv        <<<(B_*N_)/256, 256, 0, stream>>>(pf_sp, w_ds, b_ds, out);
}
